// MoEActionLayer_30090540876251
// MI455X (gfx1250) — matrix operand order repaired and checked by reading, not compiled, not run on hardware
//
#include <hip/hip_runtime.h>
#include <math.h>
#include <stdint.h>

// ---------------------------------------------------------------------------
// MoE action layer for MI455X (gfx1250, wave32, WMMA).
// Grouped (top-2 routed) execution: router -> per-expert token lists ->
// per expert: GEMM1 (4096->8192, GELU) -> GEMM2 (8192->1792, gated += out).
// All matrix math via v_wmma_f32_16x16x32_f16 (fp32 accumulate).
// Double-buffered, software-pipelined K loop (1 barrier/iter); gemm2 A-tile
// uses GLOBAL_LOAD_ASYNC_TO_LDS_B128 (ASYNCcnt) when the builtin exists.
// Deterministic: atomic slot assignment only permutes rows within tiles
// (row math is independent), expert passes are stream-serialized, final
// accumulation is plain RMW on disjoint rows per pass.
// Workspace: ~65 MB (lists + fp16 activation buffer, reused per expert).
// ---------------------------------------------------------------------------

#define HID   4096
#define FMID  8192
#define OUTD  1792
#define NEXP  4
#define NB    4096

#define TM    128     // token tile
#define TN    128     // column tile
#define KC    64      // K chunk
#define PKA   72      // LDS A pitch (halves), padded vs bank conflicts
#define PKB   72      // LDS B pitch (halves)

#if defined(__has_builtin)
#  if __has_builtin(__builtin_amdgcn_global_load_async_to_lds_b128)
#    define HAVE_ASYNC_LDS 1
#  else
#    define HAVE_ASYNC_LDS 0
#  endif
#else
#  define HAVE_ASYNC_LDS 0
#endif

typedef __attribute__((ext_vector_type(16))) _Float16 v16h;
typedef __attribute__((ext_vector_type(8)))  float    v8f;
typedef __attribute__((ext_vector_type(4)))  int      v4i;

union Frag { v16h v; uint4 q[2]; };
union H4   { _Float16 h[4]; unsigned long long u; };

__device__ __forceinline__ v8f wmma16(v16h a, v16h b, v8f c) {
  // (neg_a, A, neg_b, B, c_mod, C, reuse_a, reuse_b)
  return __builtin_amdgcn_wmma_f32_16x16x32_f16(false, a, false, b,
                                                (short)0, c, false, false);
}

__device__ __forceinline__ float gelu_exact(float x) {
  return 0.5f * x * (1.0f + erff(x * 0.70710678118654752f));
}

__device__ __forceinline__ void wait_async_zero() {
#if defined(__has_builtin) && __has_builtin(__builtin_amdgcn_s_wait_asynccnt)
  __builtin_amdgcn_s_wait_asynccnt(0);
#else
  asm volatile("s_wait_asynccnt 0" ::: "memory");
#endif
}

// ---------------------------------------------------------------------------
// Router: one wave32 per token. logits = x @ rw + rb; top-2 of 4 with
// renormalized softmax weights (softmax denominator cancels).
// ---------------------------------------------------------------------------
__global__ __launch_bounds__(128) void router_kernel(
    const float* __restrict__ hs, const float* __restrict__ rw,
    const float* __restrict__ rb, int* __restrict__ tk_idx,
    float* __restrict__ tk_w) {
  const int t    = blockIdx.x * 4 + (threadIdx.x >> 5);
  const int lane = threadIdx.x & 31;
  const float* x = hs + (size_t)t * HID;
  float a0 = 0.f, a1 = 0.f, a2 = 0.f, a3 = 0.f;
  for (int h = lane * 4; h < HID; h += 128) {
    float4 xv = *(const float4*)(x + h);
    float4 w0 = *(const float4*)(rw + (size_t)(h + 0) * NEXP);
    float4 w1 = *(const float4*)(rw + (size_t)(h + 1) * NEXP);
    float4 w2 = *(const float4*)(rw + (size_t)(h + 2) * NEXP);
    float4 w3 = *(const float4*)(rw + (size_t)(h + 3) * NEXP);
    a0 += xv.x * w0.x + xv.y * w1.x + xv.z * w2.x + xv.w * w3.x;
    a1 += xv.x * w0.y + xv.y * w1.y + xv.z * w2.y + xv.w * w3.y;
    a2 += xv.x * w0.z + xv.y * w1.z + xv.z * w2.z + xv.w * w3.z;
    a3 += xv.x * w0.w + xv.y * w1.w + xv.z * w2.w + xv.w * w3.w;
  }
  #pragma unroll
  for (int off = 16; off > 0; off >>= 1) {
    a0 += __shfl_xor(a0, off, 32);
    a1 += __shfl_xor(a1, off, 32);
    a2 += __shfl_xor(a2, off, 32);
    a3 += __shfl_xor(a3, off, 32);
  }
  if (lane == 0) {
    float l[NEXP] = {a0 + rb[0], a1 + rb[1], a2 + rb[2], a3 + rb[3]};
    int i1 = 0;
    #pragma unroll
    for (int e = 1; e < NEXP; e++) if (l[e] > l[i1]) i1 = e;   // stable ties
    int i2 = (i1 == 0) ? 1 : 0;
    #pragma unroll
    for (int e = 0; e < NEXP; e++)
      if (e != i1 && l[e] > l[i2]) i2 = e;
    float r  = __expf(l[i2] - l[i1]);   // p2/p1
    float g1 = 1.0f / (1.0f + r);
    tk_idx[t * 2 + 0] = i1;  tk_w[t * 2 + 0] = g1;
    tk_idx[t * 2 + 1] = i2;  tk_w[t * 2 + 1] = 1.0f - g1;
  }
}

// ---------------------------------------------------------------------------
// Build per-expert token lists (atomic slot assignment).
// ---------------------------------------------------------------------------
__global__ __launch_bounds__(256) void build_lists_kernel(
    const int* __restrict__ tk_idx, const float* __restrict__ tk_w,
    int* __restrict__ cnt, int* __restrict__ etok, float* __restrict__ egw) {
  int t = blockIdx.x * 256 + threadIdx.x;
  #pragma unroll
  for (int k = 0; k < 2; k++) {
    int e   = tk_idx[t * 2 + k];
    float w = tk_w[t * 2 + k];
    int pos = atomicAdd(&cnt[e], 1);
    etok[e * NB + pos] = t;
    egw[e * NB + pos]  = w;
  }
}

// ---------------------------------------------------------------------------
// GEMM1: Hbuf[slot, 0:8192] = gelu(hs[tok(slot)] @ w1[e] + b1[e])  (fp16 out)
// 128x128 tile, KC=64, 8 waves (2x4), 8 C-tiles per wave.
// Double-buffered: regs hold tile k+1 while WMMAs consume tile k.
// ---------------------------------------------------------------------------
__global__ __launch_bounds__(256) void gemm1_kernel(
    const float* __restrict__ hs, const float* __restrict__ w1,
    const float* __restrict__ b1, const int* __restrict__ etok,
    const int* __restrict__ cnt, _Float16* __restrict__ Hbuf, int e) {
  const int count = cnt[e];
  const int m0 = blockIdx.x * TM;
  if (m0 >= count) return;
  const int n0 = blockIdx.y * TN;

  __shared__ _Float16 As[2][TM * PKA];
  __shared__ _Float16 Bs[2][TN * PKB];
  __shared__ int toks[TM];

  const int tid = threadIdx.x, lane = tid & 31, wave = tid >> 5;
  const int wm = wave & 1, wn = wave >> 1;          // 2 x 4 wave grid
  const int hi = lane >> 4, lm = lane & 15;

  if (tid < TM) {
    int slot = m0 + tid;
    toks[tid] = etok[e * NB + (slot < count ? slot : count - 1)];
  }
  __syncthreads();
  const float* w1e = w1 + (size_t)e * HID * FMID;

  v8f acc[4][2];
  #pragma unroll
  for (int mt = 0; mt < 4; mt++)
    #pragma unroll
    for (int nt = 0; nt < 2; nt++) acc[mt][nt] = (v8f)(0.0f);

  auto loadA = [&](int k0, float4* r) {
    #pragma unroll
    for (int i = 0; i < 8; i++) {
      int q = i * 256 + tid;
      int m = q >> 4, kq = q & 15;                   // 16 float4 per row
      const float* src = hs + (size_t)toks[m] * HID + k0 + kq * 4;
      if (k0 + KC < HID) __builtin_prefetch(src + KC, 0, 0);
      r[i] = *(const float4*)src;
    }
  };
  auto storeA = [&](int buf, const float4* r) {
    #pragma unroll
    for (int i = 0; i < 8; i++) {
      int q = i * 256 + tid;
      int m = q >> 4, kq = q & 15;
      H4 p; p.h[0] = (_Float16)r[i].x; p.h[1] = (_Float16)r[i].y;
            p.h[2] = (_Float16)r[i].z; p.h[3] = (_Float16)r[i].w;
      *(unsigned long long*)&As[buf][m * PKA + kq * 4] = p.u;
    }
  };
  auto loadB = [&](int k0, float4* r) {
    #pragma unroll
    for (int i = 0; i < 8; i++) {
      int q = i * 256 + tid;
      int k = q >> 5, fq = q & 31;                   // 32 float4 per k-row
      const float* src = w1e + (size_t)(k0 + k) * FMID + n0 + fq * 4;
      if (k0 + KC < HID) __builtin_prefetch(src + (size_t)KC * FMID, 0, 0);
      r[i] = *(const float4*)src;
    }
  };
  auto storeB = [&](int buf, const float4* r) {
    #pragma unroll
    for (int i = 0; i < 8; i++) {
      int q = i * 256 + tid;
      int k = q >> 5, fq = q & 31;
      Bs[buf][(fq * 4 + 0) * PKB + k] = (_Float16)r[i].x;
      Bs[buf][(fq * 4 + 1) * PKB + k] = (_Float16)r[i].y;
      Bs[buf][(fq * 4 + 2) * PKB + k] = (_Float16)r[i].z;
      Bs[buf][(fq * 4 + 3) * PKB + k] = (_Float16)r[i].w;
    }
  };

  const int NIT = HID / KC;
  float4 aR[8], bR[8];
  loadA(0, aR); loadB(0, bR);
  int buf = 0;
  for (int it = 0; it < NIT; ++it) {
    storeA(buf, aR); storeB(buf, bR);
    __syncthreads();
    if (it + 1 < NIT) { loadA((it + 1) * KC, aR); loadB((it + 1) * KC, bR); }
    #pragma unroll
    for (int ks = 0; ks < KC; ks += 32) {
      Frag a[4], b[2];
      #pragma unroll
      for (int mt = 0; mt < 4; mt++) {               // A layout: lanes<16 K0..7/16..23
        int row = wm * 64 + mt * 16 + lm;
        int kb  = ks + hi * 8;
        a[mt].q[0] = *(const uint4*)&As[buf][row * PKA + kb];
        a[mt].q[1] = *(const uint4*)&As[buf][row * PKA + kb + 16];
      }
      #pragma unroll
      for (int nt = 0; nt < 2; nt++) {               // B layout: lanes<16 K0..7/16..23
        int f  = wn * 32 + nt * 16 + lm;
        int kb = ks + hi * 8;
        b[nt].q[0] = *(const uint4*)&Bs[buf][f * PKB + kb];
        b[nt].q[1] = *(const uint4*)&Bs[buf][f * PKB + kb + 16];
      }
      #pragma unroll
      for (int mt = 0; mt < 4; mt++)
        #pragma unroll
        for (int nt = 0; nt < 2; nt++)
          acc[mt][nt] = wmma16(a[mt].v, b[nt].v, acc[mt][nt]);
    }
    buf ^= 1;
  }
  // --- epilogue: bias + exact GELU, store fp16 activations ---
  #pragma unroll
  for (int nt = 0; nt < 2; nt++) {
    int f = n0 + wn * 32 + nt * 16 + lm;
    float bias = b1[(size_t)e * FMID + f];
    #pragma unroll
    for (int mt = 0; mt < 4; mt++) {
      #pragma unroll
      for (int j = 0; j < 8; j++) {                  // C layout: VGPR j -> M=j(+8)
        int slot = m0 + wm * 64 + mt * 16 + hi * 8 + j;
        float g = gelu_exact(acc[mt][nt][j] + bias);
        Hbuf[(size_t)slot * FMID + f] = (_Float16)g;
      }
    }
  }
}

// ---------------------------------------------------------------------------
// GEMM2: out[tok] += gate * (Hbuf[slot] @ w2[e] + b2[e]); disjoint rows per
// expert pass, passes are stream-serialized -> plain RMW, deterministic.
// A tile is raw fp16: copied with async global->LDS when available.
// ---------------------------------------------------------------------------
__global__ __launch_bounds__(256) void gemm2_kernel(
    const _Float16* __restrict__ Hbuf, const float* __restrict__ w2,
    const float* __restrict__ b2, const int* __restrict__ etok,
    const float* __restrict__ egw, const int* __restrict__ cnt,
    float* __restrict__ out, int e) {
  const int count = cnt[e];
  const int m0 = blockIdx.x * TM;
  if (m0 >= count) return;
  const int n0 = blockIdx.y * TN;

  __shared__ _Float16 As[2][TM * PKA];
  __shared__ _Float16 Bs[2][TN * PKB];

  const int tid = threadIdx.x, lane = tid & 31, wave = tid >> 5;
  const int wm = wave & 1, wn = wave >> 1;
  const int hi = lane >> 4, lm = lane & 15;
  const float* w2e = w2 + (size_t)e * FMID * OUTD;

  v8f acc[4][2];
  #pragma unroll
  for (int mt = 0; mt < 4; mt++)
    #pragma unroll
    for (int nt = 0; nt < 2; nt++) acc[mt][nt] = (v8f)(0.0f);

#if HAVE_ASYNC_LDS
  auto issueA = [&](int k0, int buf) {               // async global -> LDS copy
    #pragma unroll
    for (int i = 0; i < 4; i++) {
      int q = i * 256 + tid;
      int m = q >> 3, ko = q & 7;                    // 8 octs (8 halves) per row
      const _Float16* src = Hbuf + (size_t)(m0 + m) * FMID + k0 + ko * 8;
      __builtin_amdgcn_global_load_async_to_lds_b128(
          (v4i*)src, (v4i*)&As[buf][m * PKA + ko * 8], 0, 0);
    }
  };
#else
  uint4 aR[4];
  auto loadA = [&](int k0) {
    #pragma unroll
    for (int i = 0; i < 4; i++) {
      int q = i * 256 + tid;
      int m = q >> 3, ko = q & 7;
      aR[i] = *(const uint4*)(Hbuf + (size_t)(m0 + m) * FMID + k0 + ko * 8);
    }
  };
  auto storeA = [&](int buf) {
    #pragma unroll
    for (int i = 0; i < 4; i++) {
      int q = i * 256 + tid;
      int m = q >> 3, ko = q & 7;
      *(uint4*)&As[buf][m * PKA + ko * 8] = aR[i];
    }
  };
#endif
  auto loadB = [&](int k0, float4* r) {
    #pragma unroll
    for (int i = 0; i < 8; i++) {
      int q = i * 256 + tid;
      int k = q >> 5, fq = q & 31;
      const float* src = w2e + (size_t)(k0 + k) * OUTD + n0 + fq * 4;
      if (k0 + KC < FMID) __builtin_prefetch(src + (size_t)KC * OUTD, 0, 0);
      r[i] = *(const float4*)src;
    }
  };
  auto storeB = [&](int buf, const float4* r) {
    #pragma unroll
    for (int i = 0; i < 8; i++) {
      int q = i * 256 + tid;
      int k = q >> 5, fq = q & 31;
      Bs[buf][(fq * 4 + 0) * PKB + k] = (_Float16)r[i].x;
      Bs[buf][(fq * 4 + 1) * PKB + k] = (_Float16)r[i].y;
      Bs[buf][(fq * 4 + 2) * PKB + k] = (_Float16)r[i].z;
      Bs[buf][(fq * 4 + 3) * PKB + k] = (_Float16)r[i].w;
    }
  };

  const int NIT = FMID / KC;
  float4 bR[8];
#if HAVE_ASYNC_LDS
  issueA(0, 0);
#else
  loadA(0);
#endif
  loadB(0, bR);
  int buf = 0;
  for (int it = 0; it < NIT; ++it) {
    storeB(buf, bR);
#if HAVE_ASYNC_LDS
    wait_async_zero();        // our async copies into As[buf] are done
#else
    storeA(buf);
#endif
    __syncthreads();          // everyone's copies/stores visible
    if (it + 1 < NIT) {
#if HAVE_ASYNC_LDS
      issueA((it + 1) * KC, buf ^ 1);   // runs under the WMMAs below
#else
      loadA((it + 1) * KC);
#endif
      loadB((it + 1) * KC, bR);
    }
    #pragma unroll
    for (int ks = 0; ks < KC; ks += 32) {
      Frag a[4], b[2];
      #pragma unroll
      for (int mt = 0; mt < 4; mt++) {
        int row = wm * 64 + mt * 16 + lm;
        int kb  = ks + hi * 8;
        a[mt].q[0] = *(const uint4*)&As[buf][row * PKA + kb];
        a[mt].q[1] = *(const uint4*)&As[buf][row * PKA + kb + 16];
      }
      #pragma unroll
      for (int nt = 0; nt < 2; nt++) {
        int f  = wn * 32 + nt * 16 + lm;
        int kb = ks + hi * 8;
        b[nt].q[0] = *(const uint4*)&Bs[buf][f * PKB + kb];
        b[nt].q[1] = *(const uint4*)&Bs[buf][f * PKB + kb + 16];
      }
      #pragma unroll
      for (int mt = 0; mt < 4; mt++)
        #pragma unroll
        for (int nt = 0; nt < 2; nt++)
          acc[mt][nt] = wmma16(a[mt].v, b[nt].v, acc[mt][nt]);
    }
    buf ^= 1;
  }
  // --- epilogue: gated accumulate into out ---
  #pragma unroll
  for (int nt = 0; nt < 2; nt++) {
    int n = n0 + wn * 32 + nt * 16 + lm;
    float bias = b2[(size_t)e * OUTD + n];
    #pragma unroll
    for (int mt = 0; mt < 4; mt++) {
      #pragma unroll
      for (int j = 0; j < 8; j++) {
        int slot = m0 + wm * 64 + mt * 16 + hi * 8 + j;
        if (slot < count) {
          int tok = etok[e * NB + slot];
          float g = egw[e * NB + slot];
          size_t o = (size_t)tok * OUTD + n;
          out[o] += g * (acc[mt][nt][j] + bias);
        }
      }
    }
  }
}

// ---------------------------------------------------------------------------
extern "C" void kernel_launch(void* const* d_in, const int* in_sizes, int n_in,
                              void* d_out, int out_size, void* d_ws,
                              size_t ws_size, hipStream_t stream) {
  (void)in_sizes; (void)n_in; (void)ws_size;
  const float* hs = (const float*)d_in[0];
  const float* rw = (const float*)d_in[1];
  const float* rb = (const float*)d_in[2];
  const float* w1 = (const float*)d_in[3];
  const float* b1 = (const float*)d_in[4];
  const float* w2 = (const float*)d_in[5];
  const float* b2 = (const float*)d_in[6];
  float* out = (float*)d_out;

  // Workspace layout (needs ~65 MB):
  char* ws = (char*)d_ws;
  int*      cnt    = (int*)(ws + 0);                       // 4 ints
  int*      tk_idx = (int*)(ws + 256);                     // 2*NB ints
  float*    tk_w   = (float*)(ws + 256 + 32 * 1024);       // 2*NB floats
  int*      etok   = (int*)(ws + 256 + 64 * 1024);         // NEXP*NB ints
  float*    egw    = (float*)(ws + 256 + 128 * 1024);      // NEXP*NB floats
  _Float16* Hbuf   = (_Float16*)(ws + (1 << 20));          // NB*FMID fp16 = 64 MB

  (void)hipMemsetAsync(d_out, 0, (size_t)out_size * sizeof(float), stream);
  (void)hipMemsetAsync(cnt, 0, NEXP * sizeof(int), stream);

  router_kernel<<<NB / 4, 128, 0, stream>>>(hs, rw, rb, tk_idx, tk_w);
  build_lists_kernel<<<NB / 256, 256, 0, stream>>>(tk_idx, tk_w, cnt, etok, egw);

  dim3 g1(NB / TM, FMID / TN);   // 32 x 64, early-exit past count
  dim3 g2(NB / TM, OUTD / TN);   // 32 x 14
  for (int e = 0; e < NEXP; e++) {
    gemm1_kernel<<<g1, 256, 0, stream>>>(hs, w1, b1, etok, cnt, Hbuf, e);
    gemm2_kernel<<<g2, 256, 0, stream>>>(Hbuf, w2, b2, etok, egw, cnt, out, e);
  }
}
